// CRFModel_807453851948
// MI455X (gfx1250) — matrix operand order repaired and checked by reading, not compiled, not run on hardware
//
#include <hip/hip_runtime.h>
#include <hip/hip_bf16.h>
#include <math.h>

typedef __attribute__((ext_vector_type(16))) _Float16 v16h;
typedef __attribute__((ext_vector_type(8)))  _Float16 v8h;
typedef __attribute__((ext_vector_type(8)))  float    v8f;

#define NTAGS 45
#define EMB   256
#define HID   512
#define G4H   2048           // 4*HID gate dim
#define BATCH 32
#define SEQ   256
#define ROWS  (BATCH*SEQ)    // 8192
#define TPAD  48             // tags padded to 3 WMMA n-tiles
#define MT_G  (ROWS/16)      // 512 m-tiles of the flattened [S*B] dim
#define NT_G  (G4H/16)       // 128 n-tiles of the gate dim

// ---------------- workspace layout (bytes) ----------------
static constexpr size_t OFF_X    = 0;                                   // f16 [8192][256]
static constexpr size_t OFF_WIH  = OFF_X    + (size_t)ROWS*EMB*2;       // f16 [2][2048][256]
static constexpr size_t OFF_WHH  = OFF_WIH  + (size_t)2*G4H*EMB*2;      // f16 [2][2048][512]
static constexpr size_t OFF_LINW = OFF_WHH  + (size_t)2*G4H*HID*2;      // f16 [48][1024]
static constexpr size_t OFF_GIN  = OFF_LINW + (size_t)TPAD*2*HID*2;     // f16 [2][512][128][32][8] (C-tiled)
static constexpr size_t OFF_H    = OFF_GIN  + (size_t)2*ROWS*G4H*2;     // f16 [2][8192][512]
static constexpr size_t OFF_EM   = OFF_H    + (size_t)2*ROWS*HID*2;     // f32 [8192][48]

// ---------------- WMMA fragment loaders (contiguous k order, same for A and B) ----------------
// A (16x32 f16, M x K): lane L -> row M = L%16;
// K = (L/16)*16 + h, h = 0..15 (16 contiguous halves).
__device__ __forceinline__ v16h load_a_frag(const _Float16* tile, int ldk, int lane) {
  int m  = lane & 15;
  int ko = (lane >> 4) * 16;
  return *(const v16h*)(tile + m * ldk + ko);
}
// B (32x16 f16, K x N) sourced from row-major W[N][K]: lane L -> col N = L%16,
// K = (L/16)*16 + h, h = 0..15 (16 contiguous halves).
__device__ __forceinline__ v16h load_b_frag(const _Float16* tile, int ldk, int lane) {
  int n  = lane & 15;
  int ko = (lane >> 4) * 16;
  return *(const v16h*)(tile + n * ldk + ko);
}

__device__ __forceinline__ float sigf(float x) { return 1.f / (1.f + __expf(-x)); }
__device__ __forceinline__ float tanhfast(float x) {
  x = fminf(15.f, fmaxf(-15.f, x));
  float e = __expf(2.f * x);
  return (e - 1.f) / (e + 1.f);
}

// ---------------- packing kernels ----------------
__global__ void k_cast_f16(_Float16* dst, const float* src, int n) {
  int i = blockIdx.x * blockDim.x + threadIdx.x;
  if (i < n) dst[i] = (_Float16)src[i];
}

__global__ void k_pack_linw(_Float16* dst, const float* src) {
  int i = blockIdx.x * blockDim.x + threadIdx.x;      // [48][1024]
  if (i >= TPAD * 2 * HID) return;
  int r = i >> 10, c = i & 1023;
  dst[i] = (_Float16)((r < NTAGS) ? src[r * 2 * HID + c] : 0.f);
}

__global__ void k_gather_x(_Float16* X, const int* __restrict__ x,
                           const float* __restrict__ emb) {
  int i = blockIdx.x * blockDim.x + threadIdx.x;      // [8192][256], row = s*32+b
  if (i >= ROWS * EMB) return;
  int row = i >> 8, e = i & 255;
  int s = row >> 5, b = row & 31;
  int tok = x[b * SEQ + s];
  X[i] = (_Float16)emb[(size_t)tok * EMB + e];
}

// ---------------- input-projection GEMM: G = X @ W_ih^T + bias ----------------
// M=8192 (512 tiles), N=2048 (128 tiles), K=256 (8 WMMA steps). Wave = 1 mtile x 8 ntiles.
// Output stored C-fragment-tiled: one packed v8h (16 B) per lane per tile -> coalesced.
__global__ void __launch_bounds__(256) k_input_gemm(
    const _Float16* __restrict__ X, const _Float16* __restrict__ WIH,
    const float* __restrict__ bF, const float* __restrict__ bB,
    _Float16* __restrict__ GIN) {
  int dir = blockIdx.y;
  const _Float16* W = WIH + (size_t)dir * G4H * EMB;
  const float* bias = dir ? bB : bF;
  _Float16* G = GIN + (size_t)dir * ROWS * G4H;

  int lane = threadIdx.x & 31;
  int wid  = blockIdx.x * (blockDim.x >> 5) + (threadIdx.x >> 5);
  int mt = wid >> 4;        // 0..511
  int g8 = wid & 15;        // ntile group: g8*8 .. g8*8+7

  v8f z = {};
  v8f acc[8];
#pragma unroll
  for (int j = 0; j < 8; j++) acc[j] = z;

#pragma unroll
  for (int kt = 0; kt < EMB / 32; kt++) {
    v16h a = load_a_frag(X + (size_t)mt * 16 * EMB + kt * 32, EMB, lane);
    v16h bfr[8];
#pragma unroll
    for (int j = 0; j < 8; j++)           // batch all B loads -> staged loadcnt waits
      bfr[j] = load_b_frag(W + (size_t)(g8 * 8 + j) * 16 * EMB + kt * 32, EMB, lane);
#pragma unroll
    for (int j = 0; j < 8; j++)
      acc[j] = __builtin_amdgcn_wmma_f32_16x16x32_f16(false, a, false, bfr[j],
                                                      (short)0, acc[j], false, false);
  }
#pragma unroll
  for (int j = 0; j < 8; j++) {
    int ntile = g8 * 8 + j;
    float bv = bias[ntile * 16 + (lane & 15)];
    v8h pk;
#pragma unroll
    for (int r = 0; r < 8; r++) pk[r] = (_Float16)(acc[j][r] + bv);
    *(v8h*)(G + (((size_t)mt * NT_G + ntile) * 32 + lane) * 8) = pk;
  }
}

// ---------------- recurrent LSTM kernel ----------------
// grid (2 batch-halves, 2 directions), 512 threads = 16 waves.
// Wave p owns hidden units j in [32p, 32p+32): 8 WMMA tiles (4 gates x 2 j-tiles),
// so i/f/g/o for the same unit land in the same lane/VGPR -> cell update in registers.
__global__ void __launch_bounds__(512) k_recurrent(
    const _Float16* __restrict__ GIN, const _Float16* __restrict__ WHH,
    _Float16* __restrict__ H) {
  int mhalf = blockIdx.x;
  int dir   = blockIdx.y;
  const _Float16* G = GIN + (size_t)dir * ROWS * G4H;
  const _Float16* W = WHH + (size_t)dir * G4H * HID;
  _Float16* Ho = H + (size_t)dir * ROWS * HID;

  int lane = threadIdx.x & 31;
  int p    = threadIdx.x >> 5;     // 0..15
  int hi   = lane >> 4;
  int nn   = lane & 15;

  __shared__ __align__(32) _Float16 hbuf[16 * HID];   // h_{t-1}, local batch rows 0..15
  for (int i = threadIdx.x; i < 16 * HID; i += blockDim.x) hbuf[i] = (_Float16)0.f;

  float cst[2][8];                  // persistent cell state, 16 units/lane
#pragma unroll
  for (int jt = 0; jt < 2; jt++)
#pragma unroll
    for (int r = 0; r < 8; r++) cst[jt][r] = 0.f;
  __syncthreads();

  for (int step = 0; step < SEQ; step++) {
    int s = dir ? (SEQ - 1 - step) : step;
    int mtile = s * 2 + mhalf;     // row/16 of flattened [S*B]

    // warm L2/WGP$ with next timestep's precomputed gates (this wave's tiles)
    int sn = dir ? (s - 1) : (s + 1);
    if (sn >= 0 && sn < SEQ)
      __builtin_prefetch(G + (((size_t)(sn * 2 + mhalf) * NT_G + 2 * p) * 32 + lane) * 8, 0, 1);

    // C init = precomputed input projection, packed v8h per tile (coalesced b128)
    v8f acc[4][2];
#pragma unroll
    for (int g = 0; g < 4; g++)
#pragma unroll
      for (int jt = 0; jt < 2; jt++) {
        int ntile = g * 32 + 2 * p + jt;
        v8h pk = *(const v8h*)(G + (((size_t)mtile * NT_G + ntile) * 32 + lane) * 8);
#pragma unroll
        for (int r = 0; r < 8; r++) acc[g][jt][r] = (float)pk[r];
      }

    // gates += h_{t-1} @ W_hh^T  (K = 512 -> 16 WMMA steps, 8 tiles/wave)
    for (int kt = 0; kt < HID / 32; kt++) {
      v16h a = load_a_frag(hbuf + kt * 32, HID, lane);
      v16h bfr[8];
#pragma unroll
      for (int t = 0; t < 8; t++) {       // batch B loads for staged waits
        int ntile = (t >> 1) * 32 + 2 * p + (t & 1);
        bfr[t] = load_b_frag(W + (size_t)ntile * 16 * HID + kt * 32, HID, lane);
      }
#pragma unroll
      for (int t = 0; t < 8; t++)
        acc[t >> 1][t & 1] = __builtin_amdgcn_wmma_f32_16x16x32_f16(
            false, a, false, bfr[t], (short)0, acc[t >> 1][t & 1], false, false);
    }
    __syncthreads();   // everyone done reading hbuf

    // LSTM cell update entirely in registers; write new h to LDS + global
#pragma unroll
    for (int jt = 0; jt < 2; jt++) {
      int j = (2 * p + jt) * 16 + nn;
#pragma unroll
      for (int r = 0; r < 8; r++) {
        float iv = acc[0][jt][r], fv = acc[1][jt][r];
        float gv = acc[2][jt][r], ov = acc[3][jt][r];
        float c = sigf(fv) * cst[jt][r] + sigf(iv) * tanhfast(gv);
        cst[jt][r] = c;
        float h = sigf(ov) * tanhfast(c);
        int mloc = r + hi * 8;
        hbuf[mloc * HID + j] = (_Float16)h;
        Ho[((size_t)s * BATCH + mhalf * 16 + mloc) * HID + j] = (_Float16)h;
      }
    }
    __syncthreads();
  }
}

// ---------------- emissions GEMM: EM = [h_f|h_b] @ lin_w^T + lin_b ----------------
// M=8192 (512 tiles), N=48 (3 tiles), K=1024 (32 WMMA steps). Wave = 1 mtile x 3 ntiles.
__global__ void __launch_bounds__(256) k_emissions(
    const _Float16* __restrict__ H, const _Float16* __restrict__ LW,
    const float* __restrict__ lb, float* __restrict__ EM) {
  int lane = threadIdx.x & 31;
  int mt = blockIdx.x * (blockDim.x >> 5) + (threadIdx.x >> 5);   // 0..511
  const _Float16* H0 = H;
  const _Float16* H1 = H + (size_t)ROWS * HID;

  v8f z = {};
  v8f acc[3] = {z, z, z};
  for (int kt = 0; kt < 32; kt++) {
    const _Float16* Asrc = (kt < 16) ? (H0 + (size_t)mt * 16 * HID + kt * 32)
                                     : (H1 + (size_t)mt * 16 * HID + (kt - 16) * 32);
    v16h a = load_a_frag(Asrc, HID, lane);
    v16h bfr[3];
#pragma unroll
    for (int nt = 0; nt < 3; nt++)
      bfr[nt] = load_b_frag(LW + (size_t)nt * 16 * 2 * HID + kt * 32, 2 * HID, lane);
#pragma unroll
    for (int nt = 0; nt < 3; nt++)
      acc[nt] = __builtin_amdgcn_wmma_f32_16x16x32_f16(false, a, false, bfr[nt],
                                                       (short)0, acc[nt], false, false);
  }
  int hi = lane >> 4;
#pragma unroll
  for (int nt = 0; nt < 3; nt++) {
    int col = nt * 16 + (lane & 15);
    float bv = (col < NTAGS) ? lb[col] : 0.f;
#pragma unroll
    for (int r = 0; r < 8; r++)
      EM[(size_t)(mt * 16 + r + hi * 8) * TPAD + col] = acc[nt][r] + bv;
  }
}

// ---------------- CRF NLL: one block, wave w = batch b ----------------
__global__ void __launch_bounds__(1024) k_crf(
    const float* __restrict__ EM, const int* __restrict__ tags,
    const float* __restrict__ startT, const float* __restrict__ endT,
    const float* __restrict__ trans, float* __restrict__ out) {
  __shared__ float transL[NTAGS * TPAD];
  __shared__ float alphaL[BATCH][TPAD];
  __shared__ float red[BATCH];
  for (int i = threadIdx.x; i < NTAGS * NTAGS; i += blockDim.x) {
    int r = i / NTAGS, c = i % NTAGS;
    transL[r * TPAD + c] = trans[i];
  }
  __syncthreads();

  int b = threadIdx.x >> 5;
  int l = threadIdx.x & 31;
  int j0 = l;                  // always < 45
  int j1 = l + 32;
  bool v1 = (j1 < NTAGS);

  float a0 = startT[j0] + EM[(size_t)b * TPAD + j0];
  float a1 = v1 ? (startT[j1] + EM[(size_t)b * TPAD + j1]) : -1e30f;

  for (int s = 1; s < SEQ; s++) {
    alphaL[b][j0] = a0;
    if (v1) alphaL[b][j1] = a1;
    __syncthreads();
    float m0 = -1e30f, s0 = 0.f, m1 = -1e30f, s1 = 0.f;
    for (int i = 0; i < NTAGS; i++) {
      float ai = alphaL[b][i];
      float v = ai + transL[i * TPAD + j0];
      if (v > m0) { s0 = s0 * __expf(m0 - v) + 1.f; m0 = v; } else s0 += __expf(v - m0);
      if (v1) {
        float u = ai + transL[i * TPAD + j1];
        if (u > m1) { s1 = s1 * __expf(m1 - u) + 1.f; m1 = u; } else s1 += __expf(u - m1);
      }
    }
    a0 = m0 + __logf(s0) + EM[(size_t)(s * BATCH + b) * TPAD + j0];
    if (v1) a1 = m1 + __logf(s1) + EM[(size_t)(s * BATCH + b) * TPAD + j1];
    __syncthreads();
  }

  alphaL[b][j0] = a0 + endT[j0];
  if (v1) alphaL[b][j1] = a1 + endT[j1];
  __syncthreads();

  if (l == 0) {
    float m = -1e30f, ss = 0.f;
    for (int i = 0; i < NTAGS; i++) {
      float v = alphaL[b][i];
      if (v > m) { ss = ss * __expf(m - v) + 1.f; m = v; } else ss += __expf(v - m);
    }
    float logZ = m + __logf(ss);
    int prev = tags[b * SEQ];
    float sc = startT[prev];
    for (int s = 0; s < SEQ; s++) {
      int tg = tags[b * SEQ + s];
      sc += EM[(size_t)(s * BATCH + b) * TPAD + tg];
      if (s > 0) sc += trans[prev * NTAGS + tg];
      prev = tg;
    }
    sc += endT[prev];
    red[b] = logZ - sc;
  }
  __syncthreads();
  if (threadIdx.x == 0) {
    float acc = 0.f;
    for (int bb = 0; bb < BATCH; bb++) acc += red[bb];
    out[0] = acc / (float)BATCH;     // mean NLL
  }
}

// ---------------- launch ----------------
extern "C" void kernel_launch(void* const* d_in, const int* in_sizes, int n_in,
                              void* d_out, int out_size, void* d_ws, size_t ws_size,
                              hipStream_t stream) {
  (void)in_sizes; (void)n_in; (void)out_size; (void)ws_size;
  const int*   x      = (const int*)  d_in[0];
  const int*   tags   = (const int*)  d_in[1];
  const float* emb    = (const float*)d_in[2];
  const float* w_ih_f = (const float*)d_in[3];
  const float* w_hh_f = (const float*)d_in[4];
  const float* b_f    = (const float*)d_in[5];
  const float* w_ih_b = (const float*)d_in[6];
  const float* w_hh_b = (const float*)d_in[7];
  const float* b_b    = (const float*)d_in[8];
  const float* lin_w  = (const float*)d_in[9];
  const float* lin_b  = (const float*)d_in[10];
  const float* startT = (const float*)d_in[11];
  const float* endT   = (const float*)d_in[12];
  const float* trans  = (const float*)d_in[13];

  char* ws = (char*)d_ws;
  _Float16* X   = (_Float16*)(ws + OFF_X);
  _Float16* WIH = (_Float16*)(ws + OFF_WIH);
  _Float16* WHH = (_Float16*)(ws + OFF_WHH);
  _Float16* LW  = (_Float16*)(ws + OFF_LINW);
  _Float16* GIN = (_Float16*)(ws + OFF_GIN);
  _Float16* H   = (_Float16*)(ws + OFF_H);
  float*    EM  = (float*)   (ws + OFF_EM);

  const int T = 256;
  k_cast_f16 <<<(G4H * EMB + T - 1) / T, T, 0, stream>>>(WIH,             w_ih_f, G4H * EMB);
  k_cast_f16 <<<(G4H * EMB + T - 1) / T, T, 0, stream>>>(WIH + G4H * EMB, w_ih_b, G4H * EMB);
  k_cast_f16 <<<(G4H * HID + T - 1) / T, T, 0, stream>>>(WHH,             w_hh_f, G4H * HID);
  k_cast_f16 <<<(G4H * HID + T - 1) / T, T, 0, stream>>>(WHH + G4H * HID, w_hh_b, G4H * HID);
  k_pack_linw<<<(TPAD * 2 * HID + T - 1) / T, T, 0, stream>>>(LW, lin_w);
  k_gather_x <<<(ROWS * EMB + T - 1) / T, T, 0, stream>>>(X, x, emb);

  k_input_gemm<<<dim3(1024, 2), 256, 0, stream>>>(X, WIH, b_f, b_b, GIN);
  k_recurrent <<<dim3(2, 2),    512, 0, stream>>>(GIN, WHH, H);
  k_emissions <<<64,            256, 0, stream>>>(H, LW, lin_b, EM);
  k_crf       <<<1,            1024, 0, stream>>>(EM, tags, startT, endT, trans,
                                                  (float*)d_out);
}
